// FAGCN_Encoder_59425167507610
// MI455X (gfx1250) — matrix operand order repaired and checked by reading, not compiled, not run on hardware
//
#include <hip/hip_runtime.h>
#include <hip/hip_bf16.h>

typedef __attribute__((ext_vector_type(16))) _Float16 v16h;
typedef __attribute__((ext_vector_type(8)))  float    v8f;

#define N_NODES 100000
#define N_EDGES 1600000
#define DIN     128
#define HDIM    256
#define NCLS    47
#define BSZ     16384
#define NLAYER  2
#define TOPK    10
#define CHUNK   1024
#define LDLC    48   // padded leading dim for logits / transposed W2 (47 -> 48)

// ---------------- WMMA fragment loaders (wave32, per CDNA5 ISA layouts) -------------
// A 16x32 f16 fragment, lane l: M=l&15, half=l>>4.
//   elements 0..7  = A[M][kb + 8*half + e]
//   elements 8..15 = A[M][kb + 16 + 8*half + e]
// B 32x16 f16 fragment, lane l: N=l&15, half=l>>4.
//   elements 0..7  = B[kb + 8*half + e][col0+n], elements 8..15 = B[kb + 16 + 8*half + e][col0+n]
//   (we always feed B from a transposed row-major buffer Bt with Bt[n][k] = B[k][n])

union H8 { uint4 u; _Float16 h[8]; };

// A fragment from row-major f32 matrix (four aligned float4 runs, cvt to f16).
__device__ inline v16h frag_a_f32(const float* __restrict__ A, int lda, int row0, int kb, int lane) {
  int m = lane & 15, h = lane >> 4;
  const float* p = A + (size_t)(row0 + m) * lda + kb;
  float4 a0 = *(const float4*)(p + 8 * h);
  float4 a1 = *(const float4*)(p + 8 * h + 4);
  float4 a2 = *(const float4*)(p + 16 + 8 * h);
  float4 a3 = *(const float4*)(p + 16 + 8 * h + 4);
  v16h f;
  f[0]  = (_Float16)a0.x; f[1]  = (_Float16)a0.y; f[2]  = (_Float16)a0.z; f[3]  = (_Float16)a0.w;
  f[4]  = (_Float16)a1.x; f[5]  = (_Float16)a1.y; f[6]  = (_Float16)a1.z; f[7]  = (_Float16)a1.w;
  f[8]  = (_Float16)a2.x; f[9]  = (_Float16)a2.y; f[10] = (_Float16)a2.z; f[11] = (_Float16)a2.w;
  f[12] = (_Float16)a3.x; f[13] = (_Float16)a3.y; f[14] = (_Float16)a3.z; f[15] = (_Float16)a3.w;
  return f;
}

// A fragment from row-major f16 matrix (two aligned 16-byte runs).
__device__ inline v16h frag_a_f16(const _Float16* __restrict__ A, int lda, int row0, int kb, int lane) {
  int m = lane & 15, h = lane >> 4;
  const _Float16* p = A + (size_t)(row0 + m) * lda + kb;
  H8 lo, hi;
  lo.u = *(const uint4*)(p + 8 * h);
  hi.u = *(const uint4*)(p + 16 + 8 * h);
  v16h f;
#pragma unroll
  for (int e = 0; e < 8; ++e) { f[e] = lo.h[e]; f[8 + e] = hi.h[e]; }
  return f;
}

// B fragment from TRANSPOSED row-major f16 buffer Bt (Ncols x K): two aligned 16-byte runs.
__device__ inline v16h frag_bt_f16(const _Float16* __restrict__ Bt, int ldk, int col0, int kb, int lane) {
  int n = lane & 15, h = lane >> 4;
  const _Float16* p = Bt + (size_t)(col0 + n) * ldk + kb;
  H8 lo, hi;
  lo.u = *(const uint4*)(p + 8 * h);
  hi.u = *(const uint4*)(p + 16 + 8 * h);
  v16h f;
#pragma unroll
  for (int e = 0; e < 8; ++e) { f[e] = lo.h[e]; f[8 + e] = hi.h[e]; }
  return f;
}

// ---------------- Weight pre-pack kernels ------------------------------------------

// W1t[c][k] = (f16) W1[k][c] ; c in [0,HDIM), k in [0,DIN)
__global__ void k_pack_w1t(const float* __restrict__ W1, _Float16* __restrict__ W1t) {
  int i = blockIdx.x * blockDim.x + threadIdx.x;   // i = c*DIN + k
  int c = i / DIN, k = i % DIN;
  W1t[i] = (_Float16)W1[(size_t)k * HDIM + c];
}

// W2t[c][k] = (f16) W2[k][c] for c<NCLS else 0 ; c in [0,LDLC), k in [0,HDIM)
__global__ void k_pack_w2t(const float* __restrict__ W2, _Float16* __restrict__ W2t) {
  int i = blockIdx.x * blockDim.x + threadIdx.x;   // i = c*HDIM + k
  int c = i / HDIM, k = i % HDIM;
  W2t[i] = (c < NCLS) ? (_Float16)W2[(size_t)k * NCLS + c] : (_Float16)0.0f;
}

// ---------------- Main kernels -----------------------------------------------------

// h0 = relu(x @ W1 + b1). One wave per 16x64 strip (4 accumulators, A loaded once
// per k-step). grid(N/16, H/64), block 32.
__global__ void k_gemm_h0(const float* __restrict__ x, const _Float16* __restrict__ W1t,
                          const float* __restrict__ b1, float* __restrict__ h0) {
  int lane = threadIdx.x;
  int row0 = blockIdx.x * 16, col0 = blockIdx.y * 64;
  v8f c[4] = {};
#pragma unroll
  for (int kb = 0; kb < DIN; kb += 32) {
    v16h a = frag_a_f32(x, DIN, row0, kb, lane);
#pragma unroll
    for (int j = 0; j < 4; ++j) {
      v16h b = frag_bt_f16(W1t, DIN, col0 + 16 * j, kb, lane);
      c[j] = __builtin_amdgcn_wmma_f32_16x16x32_f16(false, a, false, b, (short)0, c[j], false, false);
    }
  }
  int n = lane & 15, h = lane >> 4;
#pragma unroll
  for (int j = 0; j < 4; ++j) {
    float bias = b1[col0 + 16 * j + n];
#pragma unroll
    for (int r = 0; r < 8; ++r) {
      int row = row0 + r + 8 * h;   // C/D layout: VGPR r -> M = 8*(lane/16) + r
      h0[(size_t)row * HDIM + col0 + 16 * j + n] = fmaxf(c[j][r] + bias, 0.0f);
    }
  }
}

__global__ void k_zero(float* __restrict__ p, long long n) {
  long long i = (long long)blockIdx.x * blockDim.x + threadIdx.x;
  if (i < n) p[i] = 0.0f;
}

// deg[col[e]] += 1 (float counts, scatter into dinv buffer)
__global__ void k_deg(const int* __restrict__ col, float* __restrict__ deg) {
  int e = blockIdx.x * blockDim.x + threadIdx.x;
  atomicAdd(&deg[col[e]], 1.0f);
}

__global__ void k_dinv(float* __restrict__ dinv) {
  int i = blockIdx.x * blockDim.x + threadIdx.x;
  if (i < N_NODES) {
    float d = dinv[i];
    dinv[i] = (d > 0.0f) ? rsqrtf(d) : 0.0f;
  }
}

// al = xc @ wl + bl, ar = xc @ wr + br (per-node 256-dot). One wave per node.
__global__ void k_dots(const float* __restrict__ xc, const float* __restrict__ wl,
                       const float* __restrict__ blp, const float* __restrict__ wr,
                       const float* __restrict__ brp, float* __restrict__ al,
                       float* __restrict__ ar) {
  int wid = threadIdx.x >> 5, lane = threadIdx.x & 31;
  int node = blockIdx.x * 8 + wid;
  const float* p = xc + (size_t)node * HDIM;
  float sl = 0.0f, sr = 0.0f;
#pragma unroll
  for (int j = 0; j < 8; ++j) {
    int k = lane + 32 * j;
    float v = p[k];
    sl += v * wl[k];
    sr += v * wr[k];
  }
#pragma unroll
  for (int off = 16; off; off >>= 1) {
    sl += __shfl_xor(sl, off, 32);
    sr += __shfl_xor(sr, off, 32);
  }
  if (lane == 0) { al[node] = sl + blp[0]; ar[node] = sr + brp[0]; }
}

// Edge message scatter: acc[col] += xc[row] * tanh(al[row]+ar[col]) * dinv[row]*dinv[col].
// One wave per edge, 8 f32 atomics per lane; acc (102MB) is L2-resident (192MB L2).
__global__ void k_scatter(const float* __restrict__ xc, const int* __restrict__ row,
                          const int* __restrict__ col, const float* __restrict__ al,
                          const float* __restrict__ ar, const float* __restrict__ dinv,
                          float* __restrict__ acc) {
  int wid = threadIdx.x >> 5, lane = threadIdx.x & 31;
  int e = blockIdx.x * 8 + wid;
  int r = row[e], c = col[e];
  float coef = tanhf(al[r] + ar[c]) * dinv[r] * dinv[c];
  const float* src = xc + (size_t)r * HDIM;
  float* dst = acc + (size_t)c * HDIM;
#pragma unroll
  for (int j = 0; j < 8; ++j) {
    int k = lane + 32 * j;
    atomicAdd(&dst[k], src[k] * coef);
  }
}

// xc_next = relu((EPS_ENC+EPS_CONV)*h0 + segsum), written in place over acc.
__global__ void k_update(const float* __restrict__ h0, float* __restrict__ acc) {
  size_t i = (size_t)blockIdx.x * blockDim.x + threadIdx.x;
  acc[i] = fmaxf(0.3f * h0[i] + acc[i], 0.0f);
}

// Row-normalize first B rows -> f16 `an`; also emit raw fp32 embedding output.
__global__ void k_norm_emb(const float* __restrict__ xc, _Float16* __restrict__ an,
                           float* __restrict__ emb) {
  int wid = threadIdx.x >> 5, lane = threadIdx.x & 31;
  int rowi = blockIdx.x * 8 + wid;
  const float* p = xc + (size_t)rowi * HDIM;
  float v[8]; float ss = 0.0f;
#pragma unroll
  for (int j = 0; j < 8; ++j) { v[j] = p[lane + 32 * j]; ss += v[j] * v[j]; }
#pragma unroll
  for (int off = 16; off; off >>= 1) ss += __shfl_xor(ss, off, 32);
  float scale = 1.0f / fmaxf(sqrtf(ss), 1e-8f);
#pragma unroll
  for (int j = 0; j < 8; ++j) {
    int k = lane + 32 * j;
    an[(size_t)rowi * HDIM + k]  = (_Float16)(v[j] * scale);
    emb[(size_t)rowi * HDIM + k] = v[j];
  }
}

// sim chunk: rows [rowBase, rowBase+CHUNK) x all BSZ cols. One wave per 16x64 strip
// (4 accumulators, A fragment reused 4x). B[k][n] = an[col0+n][k] (A @ A^T).
// grid(CHUNK/16, BSZ/64), block 32. 32 v_wmma per wave, 137 GFLOP total.
__global__ void k_sim(const _Float16* __restrict__ an, float* __restrict__ simc, int rowBase) {
  int lane = threadIdx.x;
  int row0 = rowBase + blockIdx.x * 16;
  int col0 = blockIdx.y * 64;
  v8f c[4] = {};
#pragma unroll
  for (int kb = 0; kb < HDIM; kb += 32) {
    v16h a = frag_a_f16(an, HDIM, row0, kb, lane);
#pragma unroll
    for (int j = 0; j < 4; ++j) {
      v16h b = frag_bt_f16(an, HDIM, col0 + 16 * j, kb, lane);
      c[j] = __builtin_amdgcn_wmma_f32_16x16x32_f16(false, a, false, b, (short)0, c[j], false, false);
    }
  }
  int n = lane & 15, h = lane >> 4;
#pragma unroll
  for (int j = 0; j < 4; ++j) {
#pragma unroll
    for (int r = 0; r < 8; ++r) {
      int lr = blockIdx.x * 16 + r + 8 * h;
      simc[(size_t)lr * BSZ + col0 + 16 * j + n] = c[j][r];
    }
  }
}

// Per-row top-10 over one sim chunk: per-thread top-10 (strided scan), LDS merge.
__global__ void k_topk(const float* __restrict__ simc, float* __restrict__ topv,
                       int* __restrict__ topi, int rowBase) {
  __shared__ float sv[256 * TOPK];
  __shared__ int   si[256 * TOPK];
  __shared__ float rv[256];
  __shared__ int   rs[256];
  int tid = threadIdx.x;
  const float* srow = simc + (size_t)blockIdx.x * BSZ;
  float tv[TOPK]; int ti[TOPK];
#pragma unroll
  for (int i = 0; i < TOPK; ++i) { tv[i] = -INFINITY; ti[i] = 0; }
  for (int j = tid; j < BSZ; j += 256) {
    float xv = srow[j];
    if (xv > tv[TOPK - 1]) {
      int p = TOPK - 1;
      while (p > 0 && xv > tv[p - 1]) { tv[p] = tv[p - 1]; ti[p] = ti[p - 1]; --p; }
      tv[p] = xv; ti[p] = j;
    }
  }
#pragma unroll
  for (int i = 0; i < TOPK; ++i) { sv[tid * TOPK + i] = tv[i]; si[tid * TOPK + i] = ti[i]; }
  __syncthreads();
  int grow = rowBase + blockIdx.x;
  for (int kk = 0; kk < TOPK; ++kk) {
    float bv = -INFINITY; int bs = 0;
#pragma unroll
    for (int q = 0; q < TOPK; ++q) {
      int s = tid * TOPK + q;
      if (sv[s] > bv) { bv = sv[s]; bs = s; }
    }
    rv[tid] = bv; rs[tid] = bs;
    __syncthreads();
    for (int off = 128; off; off >>= 1) {
      if (tid < off && rv[tid + off] > rv[tid]) { rv[tid] = rv[tid + off]; rs[tid] = rs[tid + off]; }
      __syncthreads();
    }
    if (tid == 0) {
      int s = rs[0];
      topv[(size_t)grow * TOPK + kk] = sv[s];
      topi[(size_t)grow * TOPK + kk] = si[s];
      sv[s] = -INFINITY;
    }
    __syncthreads();
  }
}

// lc = xc[:B] @ W2 (bias added later). One wave covers all 3 column tiles (48 padded
// cols) so xc rows are read once. grid(B/16), block 32.
__global__ void k_gemm_lc(const float* __restrict__ xc, const _Float16* __restrict__ W2t,
                          float* __restrict__ lc) {
  int lane = threadIdx.x;
  int row0 = blockIdx.x * 16;
  v8f c[3] = {};
#pragma unroll
  for (int kb = 0; kb < HDIM; kb += 32) {
    v16h a = frag_a_f32(xc, HDIM, row0, kb, lane);
#pragma unroll
    for (int j = 0; j < 3; ++j) {
      v16h b = frag_bt_f16(W2t, HDIM, 16 * j, kb, lane);
      c[j] = __builtin_amdgcn_wmma_f32_16x16x32_f16(false, a, false, b, (short)0, c[j], false, false);
    }
  }
  int n = lane & 15, h = lane >> 4;
#pragma unroll
  for (int j = 0; j < 3; ++j) {
#pragma unroll
    for (int r = 0; r < 8; ++r) {
      int row = row0 + r + 8 * h;
      lc[(size_t)row * LDLC + 16 * j + n] = c[j][r];
    }
  }
}

// final = 0.5*log_softmax(lc + b2) + 0.5*log_softmax(sum_k exp(topv)*onehot(y[topi]))
__global__ void k_final(const float* __restrict__ lc, const float* __restrict__ b2,
                        const float* __restrict__ topv, const int* __restrict__ topi,
                        const int* __restrict__ y, float* __restrict__ outp) {
  __shared__ float lcr[NCLS];
  __shared__ float fu[NCLS];
  __shared__ float st[2];
  int tid = threadIdx.x;
  int rowi = blockIdx.x;
  if (tid < NCLS) {
    lcr[tid] = lc[(size_t)rowi * LDLC + tid] + b2[tid];
    fu[tid] = 0.0f;
  }
  __syncthreads();
  if (tid < TOPK) {
    float v = topv[(size_t)rowi * TOPK + tid];
    int id = topi[(size_t)rowi * TOPK + tid];
    atomicAdd(&fu[y[id]], expf(v));
  }
  __syncthreads();
  if (tid == 0) {
    float m1 = -INFINITY, m2 = -INFINITY;
    for (int c = 0; c < NCLS; ++c) { m1 = fmaxf(m1, lcr[c]); m2 = fmaxf(m2, fu[c]); }
    float s1 = 0.0f, s2 = 0.0f;
    for (int c = 0; c < NCLS; ++c) { s1 += expf(lcr[c] - m1); s2 += expf(fu[c] - m2); }
    st[0] = m1 + logf(s1); st[1] = m2 + logf(s2);
  }
  __syncthreads();
  if (tid < NCLS) {
    outp[(size_t)rowi * NCLS + tid] = 0.5f * (lcr[tid] - st[0]) + 0.5f * (fu[tid] - st[1]);
  }
}

// ---------------- Host orchestration ----------------------------------------------

extern "C" void kernel_launch(void* const* d_in, const int* in_sizes, int n_in,
                              void* d_out, int out_size, void* d_ws, size_t ws_size,
                              hipStream_t stream) {
  (void)in_sizes; (void)n_in; (void)out_size; (void)ws_size;
  const float* x      = (const float*)d_in[0];
  const int*   ei     = (const int*)d_in[1];   // [row(E), col(E)]
  const int*   yv     = (const int*)d_in[2];
  const float* W1     = (const float*)d_in[4];
  const float* b1     = (const float*)d_in[5];
  const float* W2     = (const float*)d_in[6];
  const float* b2     = (const float*)d_in[7];
  const float* att_l  = (const float*)d_in[8];
  const float* att_lb = (const float*)d_in[9];
  const float* att_r  = (const float*)d_in[10];
  const float* att_rb = (const float*)d_in[11];
  float* out = (float*)d_out;

  // workspace layout
  float*    h0   = (float*)d_ws;
  float*    bufA = h0 + (size_t)N_NODES * HDIM;
  float*    bufB = bufA + (size_t)N_NODES * HDIM;
  float*    al   = bufB + (size_t)N_NODES * HDIM;
  float*    ar   = al + N_NODES;
  float*    dinv = ar + N_NODES;
  _Float16* an   = (_Float16*)(dinv + N_NODES);
  float*    simc = (float*)(an + (size_t)BSZ * HDIM);
  float*    topv = simc + (size_t)CHUNK * BSZ;
  int*      topi = (int*)(topv + (size_t)BSZ * TOPK);
  float*    lc   = (float*)(topi + (size_t)BSZ * TOPK);
  _Float16* W1t  = (_Float16*)(lc + (size_t)BSZ * LDLC);     // HDIM x DIN
  _Float16* W2t  = W1t + (size_t)HDIM * DIN;                  // LDLC x HDIM

  // 0. pack weights into transposed, padded f16 (removes all guarded/strided B loads)
  k_pack_w1t<<<(HDIM * DIN) / 256, 256, 0, stream>>>(W1, W1t);
  k_pack_w2t<<<(LDLC * HDIM) / 256, 256, 0, stream>>>(W2, W2t);

  // 1. h0 = relu(x @ W1 + b1)   (N = 6250*16 exactly; 4 col-tiles per wave)
  k_gemm_h0<<<dim3(N_NODES / 16, HDIM / 64), 32, 0, stream>>>(x, W1t, b1, h0);

  // 2. degree -> dinv
  k_zero<<<(N_NODES + 255) / 256, 256, 0, stream>>>(dinv, N_NODES);
  k_deg<<<N_EDGES / 256, 256, 0, stream>>>(ei + N_EDGES, dinv);
  k_dinv<<<(N_NODES + 255) / 256, 256, 0, stream>>>(dinv);

  // 3. FAConv layers (xc starts as h0; ping-pong acc buffers)
  const float* xc = h0;
  float* accs[2] = {bufA, bufB};
  for (int l = 0; l < NLAYER; ++l) {
    float* acc = accs[l];
    k_dots<<<N_NODES / 8, 256, 0, stream>>>(xc, att_l + l * HDIM, att_lb + l,
                                            att_r + l * HDIM, att_rb + l, al, ar);
    k_zero<<<((long long)N_NODES * HDIM + 255) / 256, 256, 0, stream>>>(
        acc, (long long)N_NODES * HDIM);
    k_scatter<<<N_EDGES / 8, 256, 0, stream>>>(xc, ei, ei + N_EDGES, al, ar, dinv, acc);
    k_update<<<(unsigned)((size_t)N_NODES * HDIM / 256), 256, 0, stream>>>(h0, acc);
    xc = acc;
  }

  // 4. embedding output + normalized f16 copy for the similarity GEMM
  k_norm_emb<<<BSZ / 8, 256, 0, stream>>>(xc, an, out + (size_t)BSZ * NCLS);

  // 5. sim = an @ an^T in 1024-row chunks (64MB scratch, L2-resident), fused top-10
  for (int cb = 0; cb < BSZ; cb += CHUNK) {
    k_sim<<<dim3(CHUNK / 16, BSZ / 64), 32, 0, stream>>>(an, simc, cb);
    k_topk<<<CHUNK, 256, 0, stream>>>(simc, topv, topi, cb);
  }

  // 6. class logits and final fused log-softmax mix
  k_gemm_lc<<<BSZ / 16, 32, 0, stream>>>(xc, W2t, lc);
  k_final<<<BSZ, 64, 0, stream>>>(lc, b2, topv, topi, yv, out);
}
